// solver_neuralSVE_85933705659120
// MI455X (gfx1250) — matrix operand order repaired and checked by reading, not compiled, not run on hardware
//
#include <hip/hip_runtime.h>

typedef __attribute__((ext_vector_type(16))) _Float16 v16h;
typedef __attribute__((ext_vector_type(8)))  _Float16 h8;
typedef __attribute__((ext_vector_type(8)))  float    v8f;

// Problem constants (match setup_inputs)
constexpr int   BATCH = 256, H = 64, D = 32, NUM = 128, KW = 32, MW = 256;
constexpr float DT    = 1.0f / (float)NUM;
constexpr int   TILE_M  = 16;
constexpr int   NBLK    = BATCH / TILE_M;   // 16 workgroups
constexpr int   THREADS = 512;              // 16 wave32 per workgroup
constexpr int   XK      = 96;               // first-layer K padded 65 -> 96
constexpr int   XSTR    = 104;              // xbuf row stride (halves), bank-spread
constexpr int   HSTR    = 264;              // hidden row stride (halves), bank-spread

// Workspace layout (bytes)
constexpr size_t OFF_KV   = 0;                              // 3*(NUM+1) f32 kernel values
constexpr size_t OFF_W1MU = 4096;                           // 48 KiB packed f16
constexpr size_t OFF_W1SG = OFF_W1MU + 49152;               // 48 KiB
constexpr size_t OFF_W2MU = OFF_W1SG + 49152;               // 32 KiB
constexpr size_t OFF_W2SG = OFF_W2MU + 32768;               // 1 MiB
constexpr size_t OFF_MUG  = OFF_W2SG + 1048576;             // 8 MiB history mu
constexpr size_t OFF_SVG  = OFF_MUG + (size_t)NBLK * NUM * TILE_M * H * 4; // 8 MiB history sv

static __device__ __forceinline__ float fast_tanh(float x) {
#if __has_builtin(__builtin_amdgcn_tanhf)
    return __builtin_amdgcn_tanhf(x);      // v_tanh_f32 on gfx1250
#else
    return tanhf(x);
#endif
}

// A-operand (16x32 f16, M x K) in this file's contiguous k order (same as packed B): lane = hf*16+ln holds row m=ln,
// halves e=0..15 map to k = k0 + 16*hf + e  -> two contiguous 16B chunks.
static __device__ __forceinline__ v16h loadA(const _Float16* rowk0, int hf) {
    h8 lo = *(const h8*)(rowk0 + 16 * hf);
    h8 hi = *(const h8*)(rowk0 + 16 * hf + 8);
    return __builtin_shufflevector(lo, hi, 0,1,2,3,4,5,6,7,8,9,10,11,12,13,14,15);
}

static __device__ __forceinline__ v8f wmma_f16(v16h a, v16h b, v8f c) {
    return __builtin_amdgcn_wmma_f32_16x16x32_f16(false, a, false, b, (short)0, c, false, false);
}

// ---------------------------------------------------------------------------
// Prep kernel 1: evaluate the three scalar kernel MLPs at all 129 time points.
// ---------------------------------------------------------------------------
__global__ void kvals_kernel(const float* __restrict__ gw1, const float* __restrict__ gb1,
                             const float* __restrict__ gw2, const float* __restrict__ gb2,
                             const float* __restrict__ mw1, const float* __restrict__ mb1,
                             const float* __restrict__ mw2, const float* __restrict__ mb2,
                             const float* __restrict__ sw1, const float* __restrict__ sb1,
                             const float* __restrict__ sw2, const float* __restrict__ sb2,
                             float* __restrict__ kv) {
    int idx = blockIdx.x * blockDim.x + threadIdx.x;
    if (idx >= 3 * (NUM + 1)) return;
    int which = idx / (NUM + 1), pt = idx % (NUM + 1);
    const float* w1 = which == 0 ? gw1 : (which == 1 ? mw1 : sw1);
    const float* b1 = which == 0 ? gb1 : (which == 1 ? mb1 : sb1);
    const float* w2 = which == 0 ? gw2 : (which == 1 ? mw2 : sw2);
    const float* b2 = which == 0 ? gb2 : (which == 1 ? mb2 : sb2);
    float t = (float)pt * DT;
    float acc = b2[0];
#pragma unroll
    for (int k = 0; k < KW; ++k)
        acc += fast_tanh(t * w1[k] + b1[k]) * w2[k];
    kv[idx] = acc;
}

// ---------------------------------------------------------------------------
// Prep kernel 2: convert f32 weight [K][N] into WMMA-B-ready packed f16 tiles.
// packed[((nt*tiles_k+kt)*32 + lane)*16 + e] = W[kt*32 + (lane>>4)*16 + e][nt*16 + (lane&15)]
// so each lane's 32 bytes per B tile are contiguous (two global_load_b128).
// ---------------------------------------------------------------------------
__global__ void pack_kernel(const float* __restrict__ W, _Float16* __restrict__ P,
                            int K, int Kpad, int N) {
    int tiles_k = Kpad / 32;
    int total = tiles_k * (N / 16) * 512;
    for (int idx = blockIdx.x * blockDim.x + threadIdx.x; idx < total;
         idx += gridDim.x * blockDim.x) {
        int e    = idx & 15;
        int lane = (idx >> 4) & 31;
        int tile = idx >> 9;
        int kt = tile % tiles_k;
        int nt = tile / tiles_k;
        int k = kt * 32 + (lane >> 4) * 16 + e;
        int n = nt * 16 + (lane & 15);
        P[idx] = (k < K) ? (_Float16)W[(size_t)k * N + n] : (_Float16)0.0f;
    }
}

// ---------------------------------------------------------------------------
// Main persistent solver: 16 workgroups, each owns a 16-row batch tile and
// runs all 128 sequential steps internally (batch rows are independent).
// ---------------------------------------------------------------------------
__global__ __launch_bounds__(THREADS) void solve_kernel(
    const float* __restrict__ Bg, const float* __restrict__ z0g,
    const float* __restrict__ mu_b1, const float* __restrict__ mu_b2,
    const float* __restrict__ sig_b1, const float* __restrict__ sig_b2,
    const _Float16* __restrict__ w1muP, const _Float16* __restrict__ w1sgP,
    const _Float16* __restrict__ w2muP, const _Float16* __restrict__ w2sgP,
    const float* __restrict__ kv, float* __restrict__ muG, float* __restrict__ svG,
    float* __restrict__ out) {
    __shared__ __attribute__((aligned(16))) _Float16 xbuf[TILE_M][XSTR];
    __shared__ __attribute__((aligned(16))) _Float16 hid[2][TILE_M][HSTR];
    __shared__ float Sbuf[2][TILE_M][H];
    __shared__ float z0s[TILE_M][H];
    __shared__ float dBls[TILE_M][D];
    __shared__ float mu_cur[TILE_M][H];
    __shared__ float sv_cur[TILE_M][H];
    __shared__ float gs[NUM + 1], kmus[NUM + 1], ksigs[NUM + 1];

    const int tid  = threadIdx.x;
    const int blk  = blockIdx.x;
    const int bm0  = blk * TILE_M;
    const int wave = tid >> 5;
    const int lane = tid & 31;
    const int ln   = lane & 15;
    const int hf   = lane >> 4;

    float* muB = muG + (size_t)blk * NUM * (TILE_M * H);
    float* svB = svG + (size_t)blk * NUM * (TILE_M * H);

    for (int idx = tid; idx < NUM + 1; idx += THREADS) {
        gs[idx]    = kv[idx];
        kmus[idx]  = kv[(NUM + 1) + idx];
        ksigs[idx] = kv[2 * (NUM + 1) + idx];
    }
    for (int idx = tid; idx < TILE_M * H; idx += THREADS) {
        int m = idx >> 6, hh = idx & 63;
        float z = z0g[(size_t)(bm0 + m) * H + hh];
        z0s[m][hh] = z;
        Sbuf[0][m][hh] = z;                    // step 0 input is z0 itself
    }
    for (int idx = tid; idx < TILE_M * (XSTR - (1 + H)); idx += THREADS) {
        int m = idx / (XSTR - (1 + H)), c = (1 + H) + idx % (XSTR - (1 + H));
        xbuf[m][c] = (_Float16)0.0f;           // zero-pad K 65..95 once
    }
    __syncthreads();
    for (int idx = tid; idx < TILE_M * H; idx += THREADS) {  // out[0] = z0 * g(0)
        int m = idx >> 6, hh = idx & 63;
        out[(size_t)(bm0 + m) * H + hh] = z0s[m][hh] * gs[0];
    }

    for (int i = 0; i < NUM; ++i) {
        // --- phase 1: build x = [t, S_i] (f16), dB_i, zero sv accumulator ---
        float tval = (float)i * DT;
        for (int idx = tid; idx < TILE_M * (1 + H); idx += THREADS) {
            int m = idx / (1 + H), c = idx % (1 + H);
            xbuf[m][c] = (c == 0) ? (_Float16)tval : (_Float16)Sbuf[i & 1][m][c - 1];
        }
        for (int idx = tid; idx < TILE_M * D; idx += THREADS) {
            int m = idx >> 5, dd = idx & 31;
            size_t base = ((size_t)(bm0 + m) * (NUM + 1) + i) * D + dd;
            dBls[m][dd] = Bg[base + D] - Bg[base];
        }
        for (int idx = tid; idx < TILE_M * H; idx += THREADS)
            (&sv_cur[0][0])[idx] = 0.0f;
        __syncthreads();

        // --- phase 2: hidden = tanh(x @ W1 + b1) for mu & sig (32 WMMA tiles) ---
        for (int t = wave; t < 32; t += 16) {
            int which = t >> 4;
            int nt = t & 15, n0 = nt * 16;
            const _Float16* wp = which ? w1sgP : w1muP;
            const float* b1p   = which ? sig_b1 : mu_b1;
            v8f acc = {};
#pragma unroll
            for (int kt = 0; kt < XK / 32; ++kt) {
                v16h a = loadA(&xbuf[ln][kt * 32], hf);
                v16h b = *(const v16h*)(wp + (((size_t)nt * (XK / 32) + kt) * 32 + lane) * 16);
                acc = wmma_f16(a, b, acc);
            }
            float bv = b1p[n0 + ln];
#pragma unroll
            for (int r = 0; r < 8; ++r) {
                int m = r + 8 * hf;
                hid[which][m][n0 + ln] = (_Float16)fast_tanh(acc[r] + bv);
            }
        }
        __syncthreads();

        // --- phase 3: output layers: 4 mu tiles + 128 sig tiles, K=256 ---
        for (int t = wave; t < 132; t += 16) {
            const bool isMu = (t < 4);
            int nt = isMu ? t : (t - 4);
            const _Float16* wp   = isMu ? w2muP : w2sgP;
            const _Float16* hrow = &hid[isMu ? 0 : 1][ln][0];
            v8f acc = {};
#pragma unroll
            for (int kt = 0; kt < MW / 32; ++kt) {
                v16h a = loadA(hrow + kt * 32, hf);
                v16h b = *(const v16h*)(wp + (((size_t)nt * (MW / 32) + kt) * 32 + lane) * 16);
                acc = wmma_f16(a, b, acc);
            }
            int n0 = nt * 16;
            if (isMu) {
                float bv = mu_b2[n0 + ln];
#pragma unroll
                for (int r = 0; r < 8; ++r) mu_cur[r + 8 * hf][n0 + ln] = acc[r] + bv;
            } else {
                // contract sig output tile with dB: column c = n0+ln -> hh=c>>5, dd=c&31
                float bv = sig_b2[n0 + ln];
                int hh = n0 >> 5, dd = (n0 & 31) + ln;
#pragma unroll
                for (int r = 0; r < 8; ++r) {
                    int m  = r + 8 * hf;
                    float v = (acc[r] + bv) * dBls[m][dd];
                    v += __shfl_xor(v, 1, 32);
                    v += __shfl_xor(v, 2, 32);
                    v += __shfl_xor(v, 4, 32);
                    v += __shfl_xor(v, 8, 32);
                    if (ln == 0) atomicAdd(&sv_cur[m][hh], v);   // ds_add_f32
                }
            }
        }
        __syncthreads();

        // --- phase 4: store history slot i; history convolution + recurrence ---
        float* muGi = muB + (size_t)i * (TILE_M * H);
        float* svGi = svB + (size_t)i * (TILE_M * H);
        for (int idx = tid; idx < TILE_M * H; idx += THREADS) {
            muGi[idx] = (&mu_cur[0][0])[idx];
            svGi[idx] = (&sv_cur[0][0])[idx];
        }
        {
            int o = tid * 2;                    // 512 threads x 2 outputs = 16*64
            int m = o >> 6, hh = o & 63;
            float am0 = 0.f, am1 = 0.f, as0 = 0.f, as1 = 0.f;
            for (int j = 0; j < i; ++j) {       // j=0 slot holds mu0/sv0
                float wm = kmus[i + 1 - j], wsg = ksigs[i + 1 - j];
                const float2 mj = *(const float2*)(muB + (size_t)j * (TILE_M * H) + o);
                const float2 sj = *(const float2*)(svB + (size_t)j * (TILE_M * H) + o);
                am0 += wm * mj.x;  am1 += wm * mj.y;
                as0 += wsg * sj.x; as1 += wsg * sj.y;
            }
            // j = i term straight from LDS (lag 1)
            am0 += kmus[1] * mu_cur[m][hh];     am1 += kmus[1] * mu_cur[m][hh + 1];
            as0 += ksigs[1] * sv_cur[m][hh];    as1 += ksigs[1] * sv_cur[m][hh + 1];
            float g1 = gs[i + 1];
            float s0 = z0s[m][hh] * g1     + DT * am0 + as0;
            float s1 = z0s[m][hh + 1] * g1 + DT * am1 + as1;
            Sbuf[(i + 1) & 1][m][hh]     = s0;
            Sbuf[(i + 1) & 1][m][hh + 1] = s1;
            size_t ob = (size_t)(i + 1) * (BATCH * H) + (size_t)(bm0 + m) * H + hh;
            out[ob]     = s0;
            out[ob + 1] = s1;
        }
        __threadfence_block();   // make this step's history stores visible next step
        __syncthreads();
    }
}

extern "C" void kernel_launch(void* const* d_in, const int* in_sizes, int n_in,
                              void* d_out, int out_size, void* d_ws, size_t ws_size,
                              hipStream_t stream) {
    (void)in_sizes; (void)n_in; (void)out_size; (void)ws_size;
    const float* Bg     = (const float*)d_in[0];
    const float* z0g    = (const float*)d_in[1];
    const float* g_w1   = (const float*)d_in[6];
    const float* g_b1   = (const float*)d_in[7];
    const float* g_w2   = (const float*)d_in[8];
    const float* g_b2   = (const float*)d_in[9];
    const float* kmu_w1 = (const float*)d_in[10];
    const float* kmu_b1 = (const float*)d_in[11];
    const float* kmu_w2 = (const float*)d_in[12];
    const float* kmu_b2 = (const float*)d_in[13];
    const float* ksg_w1 = (const float*)d_in[14];
    const float* ksg_b1 = (const float*)d_in[15];
    const float* ksg_w2 = (const float*)d_in[16];
    const float* ksg_b2 = (const float*)d_in[17];
    const float* mu_w1  = (const float*)d_in[18];
    const float* mu_b1  = (const float*)d_in[19];
    const float* mu_w2  = (const float*)d_in[20];
    const float* mu_b2  = (const float*)d_in[21];
    const float* sig_w1 = (const float*)d_in[22];
    const float* sig_b1 = (const float*)d_in[23];
    const float* sig_w2 = (const float*)d_in[24];
    const float* sig_b2 = (const float*)d_in[25];

    char* ws = (char*)d_ws;
    float*    kv    = (float*)(ws + OFF_KV);
    _Float16* w1muP = (_Float16*)(ws + OFF_W1MU);
    _Float16* w1sgP = (_Float16*)(ws + OFF_W1SG);
    _Float16* w2muP = (_Float16*)(ws + OFF_W2MU);
    _Float16* w2sgP = (_Float16*)(ws + OFF_W2SG);
    float*    muG   = (float*)(ws + OFF_MUG);
    float*    svG   = (float*)(ws + OFF_SVG);

    kvals_kernel<<<2, 256, 0, stream>>>(g_w1, g_b1, g_w2, g_b2,
                                        kmu_w1, kmu_b1, kmu_w2, kmu_b2,
                                        ksg_w1, ksg_b1, ksg_w2, ksg_b2, kv);
    pack_kernel<<<64, 256, 0, stream>>>(mu_w1,  w1muP, 65, XK, MW);
    pack_kernel<<<64, 256, 0, stream>>>(sig_w1, w1sgP, 65, XK, MW);
    pack_kernel<<<64, 256, 0, stream>>>(mu_w2,  w2muP, MW, MW, H);
    pack_kernel<<<512, 256, 0, stream>>>(sig_w2, w2sgP, MW, MW, H * D);
    solve_kernel<<<NBLK, THREADS, 0, stream>>>(Bg, z0g, mu_b1, mu_b2, sig_b1, sig_b2,
                                               w1muP, w1sgP, w2muP, w2sgP,
                                               kv, muG, svG, (float*)d_out);
}
